// ICA_GIN_62758062129644
// MI455X (gfx1250) — matrix operand order repaired and checked by reading, not compiled, not run on hardware
//
#include <hip/hip_runtime.h>
#include <hip/hip_bf16.h>

// ---------------------------------------------------------------------------
// GNN forward (GIN + GCN + decode MLP) for MI455X / gfx1250.
// Big GEMMs run on v_wmma_f32_16x16x32_f16 (wave32, one 16x16 tile per wave).
// Edge aggregations are f32 global atomics (memory-bound, ~1GB traffic).
// ---------------------------------------------------------------------------

#define N_NODES 10000
#define M_PAD   10016      // 10000 plus one spare 16-row tile
#define N_EDGES 160000
#define IN_DIM  512
#define HID     512
#define CAT     1024       // IN_DIM + HID
#define OUT_DIM 256
#define H2      128        // OUT_DIM/2

typedef __attribute__((ext_vector_type(16))) _Float16 v16h;
typedef __attribute__((ext_vector_type(8)))  _Float16 v8h;
typedef __attribute__((ext_vector_type(8)))  float    v8f;

union HV16 { v16h v; struct { v8h lo, hi; } p; };

// ---------------- element-wise converts ----------------

__global__ void cvt_x_kernel(const float* __restrict__ x, _Float16* __restrict__ xh) {
    int i = blockIdx.x * blockDim.x + threadIdx.x;           // over N*IN_DIM
    if (i >= N_NODES * IN_DIM) return;
    int m = i / IN_DIM, c = i % IN_DIM;
    xh[(size_t)m * CAT + c] = (_Float16)x[i];
}

__global__ void cvt_agg_kernel(const float* __restrict__ a, _Float16* __restrict__ ah) {
    int i = blockIdx.x * blockDim.x + threadIdx.x;           // over M_PAD*IN_DIM
    if (i >= M_PAD * IN_DIM) return;
    ah[i] = (_Float16)a[i];
}

// W[k*Nn + n] (row-major KxN) -> Wt[n*K + k] (N-major, f16)
__global__ void cvt_wT_h_kernel(const float* __restrict__ W, _Float16* __restrict__ Wt,
                                int K, int Nn) {
    int i = blockIdx.x * blockDim.x + threadIdx.x;
    if (i >= K * Nn) return;
    int n = i / K, k = i % K;
    Wt[i] = (_Float16)W[(size_t)k * Nn + n];
}

// f32 transpose for decode weight: d1_W[k*H2 + t] -> Wt[t*OUT_DIM + k]
__global__ void cvt_wT_f_kernel(const float* __restrict__ W, float* __restrict__ Wt) {
    int i = blockIdx.x * blockDim.x + threadIdx.x;           // over OUT_DIM*H2
    if (i >= OUT_DIM * H2) return;
    int t = i / OUT_DIM, k = i % OUT_DIM;
    Wt[i] = W[(size_t)k * H2 + t];
}

// ---------------- edge scatters ----------------

// GIN: agg[dst] += x[src]  (one block per edge, 128 thr * 4 floats = 512)
__global__ void gin_scatter_kernel(const float* __restrict__ x, const int* __restrict__ ei,
                                   float* __restrict__ agg) {
    const int e   = blockIdx.x;
    const int src = ei[e];
    const int dst = ei[N_EDGES + e];
    const int c   = threadIdx.x * 4;
    const float4 v = *(const float4*)(x + (size_t)src * IN_DIM + c);
    float* o = agg + (size_t)dst * IN_DIM + c;
    atomicAdd(o + 0, v.x); atomicAdd(o + 1, v.y);
    atomicAdd(o + 2, v.z); atomicAdd(o + 3, v.w);
}

__global__ void deg_kernel(const int* __restrict__ ei, float* __restrict__ deg) {
    int e = blockIdx.x * blockDim.x + threadIdx.x;
    if (e < N_EDGES) atomicAdd(&deg[ei[N_EDGES + e]], 1.0f);
}

__global__ void dis_kernel(const float* __restrict__ deg, float* __restrict__ dis) {
    int i = blockIdx.x * blockDim.x + threadIdx.x;
    if (i < N_NODES) dis[i] = rsqrtf(deg[i] + 1.0f);        // +1 = self loop
}

// GCN: out[dst] += h[src] * dis[src]*dis[dst]  (one block per edge, 64 thr * 4)
__global__ void gcn_scatter_kernel(const int* __restrict__ ei, const float* __restrict__ dis,
                                   const float* __restrict__ h, float* __restrict__ out) {
    const int e   = blockIdx.x;
    const int src = ei[e];
    const int dst = ei[N_EDGES + e];
    const float nrm = dis[src] * dis[dst];
    const int c = threadIdx.x * 4;
    const float4 v = *(const float4*)(h + (size_t)src * OUT_DIM + c);
    float* o = out + (size_t)dst * OUT_DIM + c;
    atomicAdd(o + 0, v.x * nrm); atomicAdd(o + 1, v.y * nrm);
    atomicAdd(o + 2, v.z * nrm); atomicAdd(o + 3, v.w * nrm);
}

// self-loop contribution + bias: out[i,c] += h[i,c]*dis[i]^2 + gcn_b[c]
__global__ void gcn_self_kernel(const float* __restrict__ dis, const float* __restrict__ h,
                                const float* __restrict__ gcn_b, float* __restrict__ out) {
    int i = blockIdx.x * blockDim.x + threadIdx.x;           // over N*OUT_DIM
    if (i >= N_NODES * OUT_DIM) return;
    int m = i / OUT_DIM, c = i % OUT_DIM;
    float d = dis[m];
    out[i] += h[i] * d * d + gcn_b[c];
}

// ---------------- WMMA GEMMs (one 16x16 C tile per wave32) ----------------
// A: f16 row-major [M_PAD x K], Bt: f16 N-major [Nn x K] (Bt[n*K+k] == B[k][n])

// GEMM1: h1 = agg @ f_W + f_b  -> stored f16 into xh[:, IN_DIM:CAT]
__global__ void gemm1_kernel(const _Float16* __restrict__ A, const _Float16* __restrict__ Bt,
                             const float* __restrict__ bias, _Float16* __restrict__ xh) {
    const int wave = blockIdx.x * (blockDim.x >> 5) + (threadIdx.x >> 5);
    const int lane = threadIdx.x & 31;
    const int NT = HID / 16;                                  // 32 col tiles
    const int mt = wave / NT, nt = wave % NT;
    if (mt >= M_PAD / 16) return;

    const int mA = mt * 16 + (lane & 15);
    const int nB = nt * 16 + (lane & 15);
    const int kA = (lane >= 16) ? 8 : 0;                      // A half-row split
    const int kB = (lane >= 16) ? 8 : 0;                      // B half-col split
    const _Float16* arow = A  + (size_t)mA * IN_DIM;
    const _Float16* brow = Bt + (size_t)nB * IN_DIM;

    v8f c = {};
    #pragma unroll 4
    for (int kb = 0; kb < IN_DIM; kb += 32) {
        HV16 a, b;
        a.p.lo = *(const v8h*) (arow + kb + kA);              // K = kb+kA .. +7
        a.p.hi = *(const v8h*) (arow + kb + kA + 16);         // K = kb+kA+16 .. +23
        b.p.lo = *(const v8h*) (brow + kb + kB);              // K = kb+kB .. +7
        b.p.hi = *(const v8h*) (brow + kb + kB + 16);         // K = kb+kB+16 .. +23
        c = __builtin_amdgcn_wmma_f32_16x16x32_f16(false, a.v, false, b.v,
                                                   (short)0, c, false, false);
    }
    const int   n  = nB;
    const int   mb = mt * 16 + ((lane >= 16) ? 8 : 0);
    const float bv = bias[n];
    #pragma unroll
    for (int r = 0; r < 8; ++r) {
        int m = mb + r;
        if (m < N_NODES)
            xh[(size_t)m * CAT + IN_DIM + n] = (_Float16)(c[r] + bv);
    }
}

// GEMM2: h = xh @ gcn_W (f32 out, bias added later with the aggregation)
__global__ void gemm2_kernel(const _Float16* __restrict__ A, const _Float16* __restrict__ Bt,
                             float* __restrict__ h) {
    const int wave = blockIdx.x * (blockDim.x >> 5) + (threadIdx.x >> 5);
    const int lane = threadIdx.x & 31;
    const int NT = OUT_DIM / 16;                              // 16 col tiles
    const int mt = wave / NT, nt = wave % NT;
    if (mt >= M_PAD / 16) return;

    const int mA = mt * 16 + (lane & 15);
    const int nB = nt * 16 + (lane & 15);
    const int kA = (lane >= 16) ? 8 : 0;
    const int kB = (lane >= 16) ? 8 : 0;
    const _Float16* arow = A  + (size_t)mA * CAT;
    const _Float16* brow = Bt + (size_t)nB * CAT;

    v8f c = {};
    #pragma unroll 4
    for (int kb = 0; kb < CAT; kb += 32) {
        HV16 a, b;
        a.p.lo = *(const v8h*) (arow + kb + kA);
        a.p.hi = *(const v8h*) (arow + kb + kA + 16);
        b.p.lo = *(const v8h*) (brow + kb + kB);
        b.p.hi = *(const v8h*) (brow + kb + kB + 16);
        c = __builtin_amdgcn_wmma_f32_16x16x32_f16(false, a.v, false, b.v,
                                                   (short)0, c, false, false);
    }
    const int n  = nB;
    const int mb = mt * 16 + ((lane >= 16) ? 8 : 0);
    #pragma unroll
    for (int r = 0; r < 8; ++r) {
        int m = mb + r;
        if (m < N_NODES) h[(size_t)m * OUT_DIM + n] = c[r];
    }
}

// ---------------- decode MLP: relu(out @ d1_W + d1_b) @ d2_W + d2_b ----------
__global__ void decode_kernel(const float* __restrict__ out, const float* __restrict__ d1Wt,
                              const float* __restrict__ d1b, const float* __restrict__ d2W,
                              const float* __restrict__ d2b, float* __restrict__ y) {
    __shared__ float row[OUT_DIM];
    __shared__ float zsh[H2];
    const int node = blockIdx.x;
    const int t = threadIdx.x;                                // 128 threads
    row[t]       = out[(size_t)node * OUT_DIM + t];
    row[t + H2]  = out[(size_t)node * OUT_DIM + t + H2];
    __syncthreads();

    float acc = d1b[t];
    const float4* wr = (const float4*)(d1Wt + (size_t)t * OUT_DIM);
    const float4* rr = (const float4*)row;
    #pragma unroll 8
    for (int k = 0; k < OUT_DIM / 4; ++k) {
        float4 w = wr[k], r = rr[k];
        acc += w.x * r.x + w.y * r.y + w.z * r.z + w.w * r.w;
    }
    zsh[t] = fmaxf(acc, 0.0f);
    __syncthreads();

    if (t < 2) {
        float s = d2b[t];
        #pragma unroll 4
        for (int k = 0; k < H2; ++k) s += zsh[k] * d2W[k * 2 + t];
        y[(size_t)node * 2 + t] = s;
    }
}

// ---------------------------------------------------------------------------

extern "C" void kernel_launch(void* const* d_in, const int* in_sizes, int n_in,
                              void* d_out, int out_size, void* d_ws, size_t ws_size,
                              hipStream_t stream) {
    (void)in_sizes; (void)n_in; (void)out_size; (void)ws_size;

    const float* x     = (const float*)d_in[0];
    const int*   ei    = (const int*)  d_in[1];   // [2,E]: src = ei[0:E], dst = ei[E:2E]
    const float* f_W   = (const float*)d_in[2];
    const float* f_b   = (const float*)d_in[3];
    const float* gcn_W = (const float*)d_in[4];
    const float* gcn_b = (const float*)d_in[5];
    const float* d1_W  = (const float*)d_in[6];
    const float* d1_b  = (const float*)d_in[7];
    const float* d2_W  = (const float*)d_in[8];
    const float* d2_b  = (const float*)d_in[9];
    float* y = (float*)d_out;

    // scratch carve-out
    char* ws = (char*)d_ws;
    size_t off = 0;
    auto carve = [&](size_t bytes) -> char* {
        char* p = ws + off;
        off = (off + bytes + 255) & ~(size_t)255;
        return p;
    };
    float*    agg  = (float*)   carve((size_t)M_PAD  * IN_DIM  * 4);
    _Float16* aggh = (_Float16*)carve((size_t)M_PAD  * IN_DIM  * 2);
    _Float16* xh   = (_Float16*)carve((size_t)M_PAD  * CAT     * 2);
    float*    h    = (float*)   carve((size_t)M_PAD  * OUT_DIM * 4);
    float*    outb = (float*)   carve((size_t)N_NODES* OUT_DIM * 4);
    float*    deg  = (float*)   carve((size_t)N_NODES * 4);
    float*    dis  = (float*)   carve((size_t)N_NODES * 4);
    _Float16* fWt  = (_Float16*)carve((size_t)IN_DIM * HID     * 2);
    _Float16* gWt  = (_Float16*)carve((size_t)CAT    * OUT_DIM * 2);
    float*    d1Wt = (float*)   carve((size_t)OUT_DIM* H2      * 4);

    hipMemsetAsync(agg,  0, (size_t)M_PAD   * IN_DIM  * 4, stream);
    hipMemsetAsync(deg,  0, (size_t)N_NODES * 4,           stream);
    hipMemsetAsync(outb, 0, (size_t)N_NODES * OUT_DIM * 4, stream);

    // weight / feature conversions
    cvt_wT_h_kernel<<<(IN_DIM * HID + 255) / 256, 256, 0, stream>>>(f_W, fWt, IN_DIM, HID);
    cvt_wT_h_kernel<<<(CAT * OUT_DIM + 255) / 256, 256, 0, stream>>>(gcn_W, gWt, CAT, OUT_DIM);
    cvt_wT_f_kernel<<<(OUT_DIM * H2 + 255) / 256, 256, 0, stream>>>(d1_W, d1Wt);
    cvt_x_kernel<<<(N_NODES * IN_DIM + 255) / 256, 256, 0, stream>>>(x, xh);

    // GIN aggregation + GEMM1 (h1 -> xh[:,512:])
    gin_scatter_kernel<<<N_EDGES, 128, 0, stream>>>(x, ei, agg);
    cvt_agg_kernel<<<(M_PAD * IN_DIM + 255) / 256, 256, 0, stream>>>(agg, aggh);
    gemm1_kernel<<<(M_PAD / 16) * (HID / 16) / 8, 256, 0, stream>>>(aggh, fWt, f_b, xh);

    // degrees / symmetric norm
    deg_kernel<<<(N_EDGES + 255) / 256, 256, 0, stream>>>(ei, deg);
    dis_kernel<<<(N_NODES + 255) / 256, 256, 0, stream>>>(deg, dis);

    // GCN linear + normalized aggregation
    gemm2_kernel<<<(M_PAD / 16) * (OUT_DIM / 16) / 8, 256, 0, stream>>>(xh, gWt, h);
    gcn_scatter_kernel<<<N_EDGES, 64, 0, stream>>>(ei, dis, h, outb);
    gcn_self_kernel<<<(N_NODES * OUT_DIM + 255) / 256, 256, 0, stream>>>(dis, h, gcn_b, outb);

    // decode MLP
    decode_kernel<<<N_NODES, 128, 0, stream>>>(outb, d1Wt, d1_b, d2_W, d2_b, y);
}
